// SparseDownBlock3d_75960791597673
// MI455X (gfx1250) — matrix operand order repaired and checked by reading, not compiled, not run on hardware
//
#include <hip/hip_runtime.h>
#include <hip/hip_bf16.h>
#include <math.h>

// ---------------------------------------------------------------------------
// Types for WMMA fragments (gfx1250, wave32)
// ---------------------------------------------------------------------------
typedef _Float16 half8  __attribute__((ext_vector_type(8)));
typedef _Float16 half16 __attribute__((ext_vector_type(16)));
typedef float    f32x8  __attribute__((ext_vector_type(8)));
typedef unsigned int u32x4 __attribute__((ext_vector_type(4)));

// ---------------------------------------------------------------------------
// Weight transpose + f32 -> f16 convert:  src [T][K][Nc]  ->  dst [T][Nc][K]
// ---------------------------------------------------------------------------
__global__ void wtrans_kernel(const float* __restrict__ src,
                              _Float16* __restrict__ dst,
                              int T, int K, int Nc) {
  long total = (long)T * K * Nc;
  for (long i = (long)blockIdx.x * blockDim.x + threadIdx.x; i < total;
       i += (long)gridDim.x * blockDim.x) {
    long t   = i / ((long)K * Nc);
    long rem = i - t * (long)K * Nc;
    int  kk  = (int)(rem / Nc);
    int  n   = (int)(rem - (long)kk * Nc);
    dst[(t * Nc + n) * K + kk] = (_Float16)src[i];
  }
}

// ---------------------------------------------------------------------------
// GroupNorm stats: accumulate (sum, sumsq) per group into stats[2*G]
// ---------------------------------------------------------------------------
__global__ void gn_stats_kernel(const float* __restrict__ x,
                                float* __restrict__ stats,
                                long total, int C, int chPerGroup) {
  __shared__ float s_st[64];
  if (threadIdx.x < 64) s_st[threadIdx.x] = 0.0f;
  __syncthreads();
  for (long i = (long)blockIdx.x * blockDim.x + threadIdx.x; i < total;
       i += (long)gridDim.x * blockDim.x) {
    float v = x[i];
    int c = (int)(i % C);
    int g = c / chPerGroup;
    atomicAdd(&s_st[2 * g],     v);
    atomicAdd(&s_st[2 * g + 1], v * v);
  }
  __syncthreads();
  if (threadIdx.x < 64) atomicAdd(&stats[threadIdx.x], s_st[threadIdx.x]);
}

// ---------------------------------------------------------------------------
// GN1 apply + SiLU + segment-sum pooling (h and raw x) + counts
// ---------------------------------------------------------------------------
__global__ void gn1_pool_kernel(const float* __restrict__ feats,
                                const float* __restrict__ stats,
                                const float* __restrict__ gw,
                                const float* __restrict__ gb,
                                const int*   __restrict__ seg,
                                float* __restrict__ hd_acc,
                                float* __restrict__ xd_acc,
                                float* __restrict__ cnt,
                                int Nvox, float invCount) {
  __shared__ float s_mean[32], s_inv[32];
  if (threadIdx.x < 32) {
    float m = stats[2 * threadIdx.x]     * invCount;
    float q = stats[2 * threadIdx.x + 1] * invCount;
    s_mean[threadIdx.x] = m;
    s_inv[threadIdx.x]  = rsqrtf(q - m * m + 1e-5f);
  }
  __syncthreads();
  long total = (long)Nvox * 128;
  for (long i = (long)blockIdx.x * blockDim.x + threadIdx.x; i < total;
       i += (long)gridDim.x * blockDim.x) {
    int v = (int)(i >> 7);
    int c = (int)(i & 127);
    int g = c >> 2;
    float x  = feats[i];
    float hn = (x - s_mean[g]) * s_inv[g] * gw[c] + gb[c];
    float hs = hn / (1.0f + expf(-hn));          // SiLU
    int sg = seg[v];
    atomicAdd(&xd_acc[(long)sg * 128 + c], x);
    atomicAdd(&hd_acc[(long)sg * 128 + c], hs);
    if (c == 0) atomicAdd(&cnt[sg], 1.0f);
  }
}

// ---------------------------------------------------------------------------
// Pool finalize: divide by count, convert to f16
// ---------------------------------------------------------------------------
__global__ void pool_fin_kernel(const float* __restrict__ hd_acc,
                                const float* __restrict__ xd_acc,
                                const float* __restrict__ cnt,
                                _Float16* __restrict__ hd_h,
                                _Float16* __restrict__ xd_h,
                                int M) {
  long total = (long)M * 128;
  for (long i = (long)blockIdx.x * blockDim.x + threadIdx.x; i < total;
       i += (long)gridDim.x * blockDim.x) {
    int row = (int)(i >> 7);
    float s = 1.0f / fmaxf(cnt[row], 1.0f);
    hd_h[i] = (_Float16)(hd_acc[i] * s);
    xd_h[i] = (_Float16)(xd_acc[i] * s);
  }
}

// ---------------------------------------------------------------------------
// GN2 apply + SiLU on h1 (f16 in, f16 out), C=256, G=32
// ---------------------------------------------------------------------------
__global__ void gn2_apply_kernel(const _Float16* __restrict__ h1,
                                 const float* __restrict__ stats,
                                 const float* __restrict__ gw,
                                 const float* __restrict__ gb,
                                 _Float16* __restrict__ h2,
                                 int M, float invCount) {
  __shared__ float s_mean[32], s_inv[32];
  if (threadIdx.x < 32) {
    float m = stats[2 * threadIdx.x]     * invCount;
    float q = stats[2 * threadIdx.x + 1] * invCount;
    s_mean[threadIdx.x] = m;
    s_inv[threadIdx.x]  = rsqrtf(q - m * m + 1e-5f);
  }
  __syncthreads();
  long total = (long)M * 256;
  for (long i = (long)blockIdx.x * blockDim.x + threadIdx.x; i < total;
       i += (long)gridDim.x * blockDim.x) {
    int c = (int)(i & 255);
    int g = c >> 3;
    float hv = (float)h1[i];
    float yn = (hv - s_mean[g]) * s_inv[g] * gw[c] + gb[c];
    float y  = yn / (1.0f + expf(-yn));
    h2[i] = (_Float16)y;
  }
}

// ---------------------------------------------------------------------------
// Gather-GEMM via WMMA.  One workgroup (256 thr = 8 waves) per 32-row block.
// Each wave owns a 2x2 grid of 16x16 tiles (2 row-tiles x 2 col-tiles of the
// 256 output channels), so each A/B fragment feeds two WMMAs.
// Neighbor rows are staged global->LDS with GLOBAL_LOAD_ASYNC_TO_LDS_B128
// (ASYNCcnt path); missing neighbors are zero-filled with plain LDS stores.
// MODE 0: write f16 h1 + GN2 stats;  MODE 1: store f32 out;  MODE 2: add f32 out
// ---------------------------------------------------------------------------
template <int CIN, int TAPS, bool USE_NBR, int MODE>
__global__ __launch_bounds__(256) void conv_wmma_kernel(
    const _Float16* __restrict__ xin,   // [M][CIN]  f16
    const _Float16* __restrict__ wt,    // [TAPS][256][CIN]  f16 (transposed)
    const float*    __restrict__ bias,  // [256]
    const int*      __restrict__ nbr,   // [M][27]  (USE_NBR)
    _Float16*       __restrict__ h1out, // MODE 0
    float*          __restrict__ fout,  // MODE 1/2
    float*          __restrict__ stats, // MODE 0: 64 f32
    int Mcount) {
  constexpr int ROWS = 32;
  __shared__ _Float16 a_tile[ROWS * CIN] __attribute__((aligned(16)));
  __shared__ int   s_nbr[TAPS * ROWS];   // layout [tap][row]
  __shared__ float s_stats[64];

  const int t    = threadIdx.x;
  const int lane = t & 31;
  const int wv   = t >> 5;          // 0..7
  const int row0 = blockIdx.x * ROWS;
  const int r    = lane & 15;       // fragment row / column index
  const int h    = lane >> 4;       // lane half-select

  // Stage neighbor indices: s_nbr[k*32 + rr]
  for (int j = t; j < TAPS * ROWS; j += 256) {
    int k   = j >> 5;
    int rr  = j & 31;
    int row = row0 + rr;
    int idx = -1;
    if (row < Mcount) idx = USE_NBR ? nbr[(long)row * 27 + k] : row;
    s_nbr[j] = idx;
  }
  if (MODE == 0 && t < 64) s_stats[t] = 0.0f;
  __syncthreads();

  f32x8 acc[2][2] = {};
  const int n0 = wv * 32;
  const int n1 = n0 + 16;

  for (int k = 0; k < TAPS; ++k) {
    // ---- gather 32 neighbor rows into LDS via async B128 copies
    constexpr int CHUNKS = CIN / 8;            // 16B chunks per row (16 or 32)
    for (int j = t; j < ROWS * CHUNKS; j += 256) {
      int rr  = j / CHUNKS;                    // power-of-two divide
      int c   = j - rr * CHUNKS;
      int idx = s_nbr[k * ROWS + rr];
      _Float16* dst = &a_tile[rr * CIN + c * 8];
      if (idx >= 0) {
        unsigned long long ga =
            (unsigned long long)(const void*)(xin + (long)idx * CIN + c * 8);
        unsigned lds = (unsigned)(unsigned long long)(void*)dst;
        asm volatile("global_load_async_to_lds_b128 %0, %1, off"
                     :: "v"(lds), "v"(ga) : "memory");
      } else {
        u32x4 z = {0u, 0u, 0u, 0u};
        *(u32x4*)dst = z;
      }
    }
    asm volatile("s_wait_asynccnt 0x0" ::: "memory");
    __syncthreads();

    const _Float16* wk = wt + (long)k * 256 * CIN;
    if (k + 1 < TAPS) {  // hint next tap's weight rows into cache
      const _Float16* wn = wt + (long)(k + 1) * 256 * CIN;
      __builtin_prefetch(wn + (long)(n0 + r) * CIN, 0, 1);
      __builtin_prefetch(wn + (long)(n1 + r) * CIN, 0, 1);
    }

#pragma unroll
    for (int ks = 0; ks < CIN / 32; ++ks) {
      const int kb = ks * 32;
      // A fragments (16x32 f16): lanes 0-15 hold
      // K[kb..kb+15], lanes 16-31 K[kb+16..kb+31] (contiguous, matches B)
      half16 a0, a1;
      {
        half8 lo = *(const half8*)(&a_tile[r * CIN + kb + h * 16]);
        half8 hi = *(const half8*)(&a_tile[r * CIN + kb + h * 16 + 8]);
#pragma unroll
        for (int e = 0; e < 8; ++e) { a0[e] = lo[e]; a0[8 + e] = hi[e]; }
      }
      {
        half8 lo = *(const half8*)(&a_tile[(16 + r) * CIN + kb + h * 16]);
        half8 hi = *(const half8*)(&a_tile[(16 + r) * CIN + kb + h * 16 + 8]);
#pragma unroll
        for (int e = 0; e < 8; ++e) { a1[e] = lo[e]; a1[8 + e] = hi[e]; }
      }
      // B fragments (32x16 f16): lane column = r, K = h*16 + 0..15 contiguous
      half16 b0 = *(const half16*)(wk + (long)(n0 + r) * CIN + kb + h * 16);
      half16 b1 = *(const half16*)(wk + (long)(n1 + r) * CIN + kb + h * 16);
      acc[0][0] = __builtin_amdgcn_wmma_f32_16x16x32_f16(false, a0, false, b0,
                                                         (short)0, acc[0][0], false, false);
      acc[0][1] = __builtin_amdgcn_wmma_f32_16x16x32_f16(false, a0, false, b1,
                                                         (short)0, acc[0][1], false, false);
      acc[1][0] = __builtin_amdgcn_wmma_f32_16x16x32_f16(false, a1, false, b0,
                                                         (short)0, acc[1][0], false, false);
      acc[1][1] = __builtin_amdgcn_wmma_f32_16x16x32_f16(false, a1, false, b1,
                                                         (short)0, acc[1][1], false, false);
    }
    __syncthreads();   // protect a_tile before next tap's gather
  }

  // ---- epilogue: D layout -> lane column = r, VGPR v -> row rt*16 + h*8 + v
  const int c0 = n0 + r, c1 = n1 + r;
  const float b0s = bias[c0], b1s = bias[c1];
  if (MODE == 0) {
    float s0 = 0, q0 = 0, s1 = 0, q1 = 0;
#pragma unroll
    for (int rt = 0; rt < 2; ++rt) {
#pragma unroll
      for (int v = 0; v < 8; ++v) {
        int row = row0 + rt * 16 + h * 8 + v;
        if (row < Mcount) {
          float y0 = acc[rt][0][v] + b0s;
          float y1 = acc[rt][1][v] + b1s;
          h1out[(long)row * 256 + c0] = (_Float16)y0;
          h1out[(long)row * 256 + c1] = (_Float16)y1;
          s0 += y0; q0 += y0 * y0;
          s1 += y1; q1 += y1 * y1;
        }
      }
    }
    atomicAdd(&s_stats[(c0 >> 3) * 2],     s0);
    atomicAdd(&s_stats[(c0 >> 3) * 2 + 1], q0);
    atomicAdd(&s_stats[(c1 >> 3) * 2],     s1);
    atomicAdd(&s_stats[(c1 >> 3) * 2 + 1], q1);
    __syncthreads();
    if (t < 64) atomicAdd(&stats[t], s_stats[t]);
  } else {
#pragma unroll
    for (int rt = 0; rt < 2; ++rt) {
#pragma unroll
      for (int v = 0; v < 8; ++v) {
        int row = row0 + rt * 16 + h * 8 + v;
        if (row < Mcount) {
          long i0 = (long)row * 256 + c0;
          long i1 = (long)row * 256 + c1;
          float y0 = acc[rt][0][v] + b0s;
          float y1 = acc[rt][1][v] + b1s;
          if (MODE == 2) { y0 += fout[i0]; y1 += fout[i1]; }
          fout[i0] = y0;
          fout[i1] = y1;
        }
      }
    }
  }
}

// ---------------------------------------------------------------------------
// Host launcher
// ---------------------------------------------------------------------------
extern "C" void kernel_launch(void* const* d_in, const int* in_sizes, int n_in,
                              void* d_out, int out_size, void* d_ws, size_t ws_size,
                              hipStream_t stream) {
  const float* feats  = (const float*)d_in[0];
  const float* gn1_w  = (const float*)d_in[1];
  const float* gn1_b  = (const float*)d_in[2];
  const float* W1     = (const float*)d_in[3];
  const float* b1     = (const float*)d_in[4];
  const float* gn2_w  = (const float*)d_in[5];
  const float* gn2_b  = (const float*)d_in[6];
  const float* W2     = (const float*)d_in[7];
  const float* b2     = (const float*)d_in[8];
  const float* Wskip  = (const float*)d_in[9];
  const float* bskip  = (const float*)d_in[10];
  const int*   seg    = (const int*)d_in[11];
  const int*   nbr    = (const int*)d_in[12];

  const int N = in_sizes[0] / 128;     // voxels at full res
  const int M = in_sizes[12] / 27;     // downsampled voxels

  // ---- workspace layout
  char* ws = (char*)d_ws;
  size_t off = 0;
  auto take = [&](size_t bytes) -> char* {
    char* p = ws + off;
    off = (off + bytes + 255) & ~(size_t)255;
    return p;
  };
  float*    statsA = (float*)take(64 * sizeof(float));
  float*    statsB = (float*)take(64 * sizeof(float));
  float*    cnt    = (float*)take((size_t)M * sizeof(float));
  float*    hd_acc = (float*)take((size_t)M * 128 * sizeof(float));
  float*    xd_acc = (float*)take((size_t)M * 128 * sizeof(float));
  _Float16* hd_h   = (_Float16*)take((size_t)M * 128 * sizeof(_Float16));
  _Float16* xd_h   = (_Float16*)take((size_t)M * 128 * sizeof(_Float16));
  _Float16* h1_h   = (_Float16*)take((size_t)M * 256 * sizeof(_Float16));
  _Float16* h2_h   = (_Float16*)take((size_t)M * 256 * sizeof(_Float16));
  _Float16* W1t    = (_Float16*)take((size_t)27 * 256 * 128 * sizeof(_Float16));
  _Float16* W2t    = (_Float16*)take((size_t)27 * 256 * 256 * sizeof(_Float16));
  _Float16* Wst    = (_Float16*)take((size_t)256 * 128 * sizeof(_Float16));
  (void)ws_size; (void)n_in; (void)out_size;

  // ---- zero accumulators (graph-capture safe)
  hipMemsetAsync(statsA, 0, 64 * sizeof(float), stream);
  hipMemsetAsync(statsB, 0, 64 * sizeof(float), stream);
  hipMemsetAsync(cnt,    0, (size_t)M * sizeof(float), stream);
  hipMemsetAsync(hd_acc, 0, (size_t)M * 128 * sizeof(float), stream);
  hipMemsetAsync(xd_acc, 0, (size_t)M * 128 * sizeof(float), stream);

  // ---- weight transpose/convert to f16 [tap][Cout][Cin]
  {
    long t1 = 27L * 128 * 256, t2 = 27L * 256 * 256, ts = 128L * 256;
    wtrans_kernel<<<(int)((t1 + 255) / 256), 256, 0, stream>>>(W1, W1t, 27, 128, 256);
    wtrans_kernel<<<(int)((t2 + 255) / 256), 256, 0, stream>>>(W2, W2t, 27, 256, 256);
    wtrans_kernel<<<(int)((ts + 255) / 256), 256, 0, stream>>>(Wskip, Wst, 1, 128, 256);
  }

  // ---- GN1 stats, apply + pool, finalize
  gn_stats_kernel<<<2048, 256, 0, stream>>>(feats, statsA, (long)N * 128, 128, 4);
  gn1_pool_kernel<<<4096, 256, 0, stream>>>(feats, statsA, gn1_w, gn1_b, seg,
                                            hd_acc, xd_acc, cnt, N,
                                            1.0f / ((float)N * 4.0f));
  pool_fin_kernel<<<2048, 256, 0, stream>>>(hd_acc, xd_acc, cnt, hd_h, xd_h, M);

  const int mtiles = (M + 31) / 32;

  // ---- conv1: 27 taps, 128->256, fused bias + GN2 stats, f16 out
  conv_wmma_kernel<128, 27, true, 0><<<mtiles, 256, 0, stream>>>(
      hd_h, W1t, b1, nbr, h1_h, nullptr, statsB, M);

  // ---- GN2 + SiLU
  gn2_apply_kernel<<<2048, 256, 0, stream>>>(h1_h, statsB, gn2_w, gn2_b, h2_h,
                                             M, 1.0f / ((float)M * 8.0f));

  // ---- skip: 1 tap identity, 128->256, writes f32 d_out
  conv_wmma_kernel<128, 1, false, 1><<<mtiles, 256, 0, stream>>>(
      xd_h, Wst, bskip, nullptr, nullptr, (float*)d_out, nullptr, M);

  // ---- conv2: 27 taps, 256->256, adds into d_out
  conv_wmma_kernel<256, 27, true, 2><<<mtiles, 256, 0, stream>>>(
      h2_h, W2t, b2, nbr, nullptr, (float*)d_out, nullptr, M);
}
